// Pointnet2Backbone_62955630624802
// MI455X (gfx1250) — matrix operand order repaired and checked by reading, not compiled, not run on hardware
//
#include <hip/hip_runtime.h>
#include <hip/hip_bf16.h>

typedef __attribute__((ext_vector_type(16))) _Float16 v16h;
typedef __attribute__((ext_vector_type(8)))  _Float16 v8h;
typedef __attribute__((ext_vector_type(8)))  float    v8f;

#define EPSBN 1e-5f

// ---- WMMA 16x16x32 f16 layouts (CDNA5 ISA 7.12.2) ----
// A 16x32 f16: lane l holds row l&15; elems 0-7 -> K = (l>>4)*8 + 0..7, elems 8-15 -> +16.
// B 32x16 f16: lane l holds col l&15; code packs A and B alike: elem e -> K = (l>>4)*16 + e.
// C/D f32: lane l holds col l&15; vgpr j -> row j + 8*(l>>4).

__device__ __forceinline__ v16h cvt16(float4 a0, float4 a1, float4 a2, float4 a3) {
    v16h v;
    v[0]=(_Float16)a0.x; v[1]=(_Float16)a0.y; v[2]=(_Float16)a0.z; v[3]=(_Float16)a0.w;
    v[4]=(_Float16)a1.x; v[5]=(_Float16)a1.y; v[6]=(_Float16)a1.z; v[7]=(_Float16)a1.w;
    v[8]=(_Float16)a2.x; v[9]=(_Float16)a2.y; v[10]=(_Float16)a2.z; v[11]=(_Float16)a2.w;
    v[12]=(_Float16)a3.x; v[13]=(_Float16)a3.y; v[14]=(_Float16)a3.z; v[15]=(_Float16)a3.w;
    return v;
}

// =====================================================================
// Generic GEMM: C[M,N] = act( A[M,K] @ W (+bias) (+resid) )
// REQUIRES: K % 32 == 0, M % 64 == 0, N % 16 == 0  (true at every call site)
// WT=1 -> W stored [N,K] (x @ W.T); WT=0 -> W stored [K,N]
// ACT: 0 none, 1 leaky_relu(0.01). RESID: add resid before act.
// block = 128 (4 waves = 4 consecutive row tiles), grid = (N/16, M/64).
// =====================================================================
template <int WT, bool RESID, int ACT>
__global__ void gemm_wmma_kernel(const float* __restrict__ A, const float* __restrict__ W,
                                 const float* __restrict__ bias, const float* __restrict__ resid,
                                 float* __restrict__ C, int M, int K, int N)
{
    int lane = threadIdx.x & 31;
    int wave = threadIdx.x >> 5;
    int tm = blockIdx.y * 4 + wave;
    int tn = blockIdx.x;

    int rowA = tm * 16 + (lane & 15);
    int col  = tn * 16 + (lane & 15);
    int grp8  = (lane >> 4) << 3;   // 0 or 8
    int grp16 = (lane >> 4) << 4;   // 0 or 16

    const float* arow = A + (size_t)rowA * K + grp16;

    v8f acc = {};
#pragma unroll 2
    for (int kb = 0; kb < K; kb += 32) {
        const float* ap = arow + kb;
        v16h av = cvt16(*(const float4*)(ap + 0),  *(const float4*)(ap + 4),
                        *(const float4*)(ap + 8), *(const float4*)(ap + 12));
        v16h bv;
        if (WT) {
            const float* wp = W + (size_t)col * K + kb + grp16;   // 16 consecutive floats
            bv = cvt16(*(const float4*)(wp + 0), *(const float4*)(wp + 4),
                       *(const float4*)(wp + 8), *(const float4*)(wp + 12));
        } else {
            const float* wp = W + (size_t)(kb + grp16) * N + col; // stride-N column
#pragma unroll
            for (int e = 0; e < 16; ++e)
                bv[e] = (_Float16)wp[(size_t)e * N];
        }
        acc = __builtin_amdgcn_wmma_f32_16x16x32_f16(false, av, false, bv, (short)0, acc, false, false);
    }

    float bb = bias[col];
#pragma unroll
    for (int j = 0; j < 8; ++j) {
        int row = tm * 16 + grp8 + j;
        float v = acc[j] + bb;
        if (RESID) v += resid[(size_t)row * N + col];
        if (ACT == 1) v = (v > 0.0f) ? v : 0.01f * v;
        C[(size_t)row * N + col] = v;
    }
}

// =====================================================================
// fc0: f = leaky_relu(bn(features @ W0 + b0))   features (B*N,3), W0 (3,16)
// =====================================================================
__global__ void fc0_kernel(const float* __restrict__ feat, const float* __restrict__ w,
                           const float* __restrict__ b, const float* __restrict__ g,
                           const float* __restrict__ bb, const float* __restrict__ mm,
                           const float* __restrict__ vv, float* __restrict__ fout, int total)
{
    int i = blockIdx.x * blockDim.x + threadIdx.x;
    if (i >= total) return;
    float x0 = feat[(size_t)i * 3 + 0];
    float x1 = feat[(size_t)i * 3 + 1];
    float x2 = feat[(size_t)i * 3 + 2];
#pragma unroll
    for (int c = 0; c < 16; ++c) {
        float y = x0 * w[c] + x1 * w[16 + c] + x2 * w[32 + c] + b[c];
        y = (y - mm[c]) * rsqrtf(vv[c] + EPSBN) * g[c] + bb[c];
        y = (y > 0.0f) ? y : 0.01f * y;
        fout[(size_t)i * 16 + c] = y;
    }
}

// =====================================================================
// Farthest point sampling: one block (256 thr) per batch; jax-scan semantics
// (emit sel, update dist, first-max argmax).
// =====================================================================
__global__ void fps_kernel(const float* __restrict__ pc, int Nc, int npoint,
                           float* __restrict__ dist, int* __restrict__ outIdx)
{
    int b = blockIdx.x;
    const float* xyz = pc + (size_t)b * Nc * 3;
    float* d = dist + (size_t)b * Nc;
    __shared__ float sv[256];
    __shared__ int   si[256];
    __shared__ float spt[3];
    __shared__ int   ssel;
    int t = threadIdx.x;
    for (int i = t; i < Nc; i += blockDim.x) d[i] = 1e10f;
    if (t == 0) ssel = 0;
    __syncthreads();
    for (int it = 0; it < npoint; ++it) {
        int sel = ssel;
        if (t == 0) {
            outIdx[(size_t)b * npoint + it] = sel;
            spt[0] = xyz[(size_t)sel * 3 + 0];
            spt[1] = xyz[(size_t)sel * 3 + 1];
            spt[2] = xyz[(size_t)sel * 3 + 2];
        }
        __syncthreads();
        float px = spt[0], py = spt[1], pz = spt[2];
        float best = -1.0f; int bi = 0;
        for (int i = t; i < Nc; i += blockDim.x) {
            float dx = xyz[(size_t)i * 3 + 0] - px;
            float dy = xyz[(size_t)i * 3 + 1] - py;
            float dz = xyz[(size_t)i * 3 + 2] - pz;
            float dd = dx * dx + dy * dy + dz * dz;
            float nd = fminf(d[i], dd);
            d[i] = nd;
            if (nd > best) { best = nd; bi = i; }
        }
        sv[t] = best; si[t] = bi;
        __syncthreads();
        for (int off = 128; off > 0; off >>= 1) {
            if (t < off) {
                if (sv[t + off] > sv[t] || (sv[t + off] == sv[t] && si[t + off] < si[t])) {
                    sv[t] = sv[t + off]; si[t] = si[t + off];
                }
            }
            __syncthreads();
        }
        if (t == 0) ssel = si[0];
        __syncthreads();
    }
}

__global__ void gather_xyz_kernel(const float* __restrict__ pc, const int* __restrict__ idx,
                                  float* __restrict__ out, int Nc, int npoint, int total)
{
    int gid = blockIdx.x * blockDim.x + threadIdx.x;
    if (gid >= total) return;
    int b = gid / npoint;
    int p = idx[gid];
    const float* src = pc + ((size_t)b * Nc + p) * 3;
    out[(size_t)gid * 3 + 0] = src[0];
    out[(size_t)gid * 3 + 1] = src[1];
    out[(size_t)gid * 3 + 2] = src[2];
}

// =====================================================================
// Ball query: one wave32 per query; ballot gathers first nsample in-radius
// indices in ascending order (== top-k-of-index semantics); pad with first.
// =====================================================================
__global__ void ball_query_kernel(const float* __restrict__ pc, const float* __restrict__ nxyz,
                                  int Nc, int npoint, int nsample, float r2,
                                  int* __restrict__ outIdx, int qtot)
{
    int q = blockIdx.x * (blockDim.x >> 5) + (threadIdx.x >> 5);
    int lane = threadIdx.x & 31;
    if (q >= qtot) return;                        // uniform per wave
    int b = q / npoint;
    const float* xyz = pc + (size_t)b * Nc * 3;
    float cx = nxyz[(size_t)q * 3 + 0];
    float cy = nxyz[(size_t)q * 3 + 1];
    float cz = nxyz[(size_t)q * 3 + 2];
    int* out = outIdx + (size_t)q * nsample;
    int count = 0, first = -1;
    for (int base = 0; base < Nc; base += 32) {
        int i = base + lane;
        bool in = false;
        if (i < Nc) {
            float dx = xyz[(size_t)i * 3 + 0] - cx;
            float dy = xyz[(size_t)i * 3 + 1] - cy;
            float dz = xyz[(size_t)i * 3 + 2] - cz;
            in = (dx * dx + dy * dy + dz * dz) < r2;
        }
        unsigned mask = __builtin_amdgcn_ballot_w32(in);
        if (in) {
            int pos = count + __popc(mask & ((1u << lane) - 1u));
            if (pos < nsample) out[pos] = i;
        }
        if (first < 0 && mask) first = base + __ffs(mask) - 1;
        count += __popc(mask);
        if (count >= nsample) break;
    }
    if (first < 0) first = 0;
    if (lane == 0) {
        int c2 = (count < nsample) ? count : nsample;
        for (int p = c2; p < nsample; ++p) out[p] = first;
    }
}

// =====================================================================
// Fused SA layer: gather tile -> LDS (f16) -> WMMA -> BN -> ReLU -> max.
// One block per (b, s); waves = (nsample/16)*(co/16).
// Full K chunks are clamp-free; single ragged tail chunk uses clamped
// addresses + value select (no exec-mask branching).
// =====================================================================
__global__ void sa_conv_kernel(const float* __restrict__ pc, const float* __restrict__ nxyz,
                               const float* __restrict__ feat, const int* __restrict__ ballIdx,
                               const float* __restrict__ W, const float* __restrict__ g,
                               const float* __restrict__ bb, const float* __restrict__ mm,
                               const float* __restrict__ vv, float* __restrict__ fout,
                               int Nprev, int npoint, int nsample, int ci, int co)
{
    __shared__ _Float16 Atile[32 * 288];   // max nsample=32 x Kpad<=288
    __shared__ int smax[256];
    int bs = blockIdx.x;
    int b = bs / npoint;
    int wave = threadIdx.x >> 5;
    int lane = threadIdx.x & 31;
    int nwaves = blockDim.x >> 5;
    int colTiles = co >> 4;
    int rt = wave / colTiles, ct = wave % colTiles;

    for (int i = threadIdx.x; i < co; i += blockDim.x) smax[i] = 0;

    int K = ci + 3;
    int Kpad = (K + 31) & ~31;
    int Kfull = K & ~31;                    // clamp-free chunks
    const int* idxRow = ballIdx + (size_t)bs * nsample;
    float cx = nxyz[(size_t)bs * 3 + 0];
    float cy = nxyz[(size_t)bs * 3 + 1];
    float cz = nxyz[(size_t)bs * 3 + 2];

    // ---- cooperative gather into LDS: one wave per sample row ----
    for (int r = wave; r < nsample; r += nwaves) {
        int pidx = idxRow[r];
        const float* pr = pc + ((size_t)b * Nprev + pidx) * 3;
        const float* fr = feat + ((size_t)b * Nprev + pidx) * ci;
        float gx = pr[0] - cx, gy = pr[1] - cy, gz = pr[2] - cz;
        for (int k = lane; k < Kpad; k += 32) {
            int kf = k - 3;
            kf = (kf < 0) ? 0 : (kf >= ci ? ci - 1 : kf);
            float fv = fr[kf];                       // unconditional clamped load
            float v = (k == 0) ? gx : (k == 1) ? gy : (k == 2) ? gz : fv;
            v = (k < K) ? v : 0.0f;
            Atile[(size_t)r * Kpad + k] = (_Float16)v;
        }
    }
    __syncthreads();

    int nIdx = rt * 16 + (lane & 15);
    int col  = ct * 16 + (lane & 15);
    int grp8  = (lane >> 4) << 3;
    int grp16 = (lane >> 4) << 4;
    const _Float16* arow = Atile + (size_t)nIdx * Kpad + grp16;

    v8f acc = {};
    // full chunks: no clamps
    for (int kb = 0; kb < Kfull; kb += 32) {
        v8h lo = *(const v8h*)(arow + kb);          // ds_load_b128
        v8h hi = *(const v8h*)(arow + kb + 8);
        v16h av = __builtin_shufflevector(lo, hi, 0,1,2,3,4,5,6,7,8,9,10,11,12,13,14,15);
        const float* wp = W + (size_t)(kb + grp16) * co + col;
        v16h bv;
#pragma unroll
        for (int e = 0; e < 16; ++e)
            bv[e] = (_Float16)wp[(size_t)e * co];
        acc = __builtin_amdgcn_wmma_f32_16x16x32_f16(false, av, false, bv, (short)0, acc, false, false);
    }
    // ragged tail chunk (K % 32 != 0 for all stages)
    {
        int kb = Kfull;
        v8h lo = *(const v8h*)(arow + kb);
        v8h hi = *(const v8h*)(arow + kb + 8);
        v16h av = __builtin_shufflevector(lo, hi, 0,1,2,3,4,5,6,7,8,9,10,11,12,13,14,15);
        v16h bv;
#pragma unroll
        for (int e = 0; e < 16; ++e) {
            int kw = kb + grp16 + e;
            int kc = (kw < K) ? kw : K - 1;          // clamped address
            float fw = W[(size_t)kc * co + col];     // unconditional load
            bv[e] = (_Float16)((kw < K) ? fw : 0.0f);
        }
        acc = __builtin_amdgcn_wmma_f32_16x16x32_f16(false, av, false, bv, (short)0, acc, false, false);
    }

    float gg = g[col], bbn = bb[col], mmn = mm[col];
    float inv = rsqrtf(vv[col] + EPSBN);
    float mx = 0.0f;
#pragma unroll
    for (int j = 0; j < 8; ++j) {
        float h = (acc[j] - mmn) * inv * gg + bbn;
        h = (h > 0.0f) ? h : 0.0f;
        mx = fmaxf(mx, h);
    }
    atomicMax(&smax[col], __float_as_int(mx));       // post-ReLU >= 0: int-bit max valid
    __syncthreads();
    for (int i = threadIdx.x; i < co; i += blockDim.x)
        fout[(size_t)bs * co + i] = __int_as_float(smax[i]);
}

// =====================================================================
// Attention core: one thread per (b, head, query); online softmax. hd <= 32.
// =====================================================================
__global__ void attn_kernel(const float* __restrict__ qkv, float* __restrict__ obuf,
                            int S, int D, int hd, int total)
{
    int gid = blockIdx.x * blockDim.x + threadIdx.x;
    if (gid >= total) return;
    int qi = gid % S;
    int h  = (gid / S) & 7;
    int b  = gid / (S * 8);
    const float* base = qkv + (size_t)b * S * 3 * D;
    const float* qrow = base + (size_t)qi * 3 * D + h * hd;
    float q[32], o[32];
    for (int d = 0; d < hd; ++d) { q[d] = qrow[d]; o[d] = 0.0f; }
    float scale = 1.0f / sqrtf((float)hd);
    float m = -1e30f, ssum = 0.0f;
    for (int j = 0; j < S; ++j) {
        const float* krow = base + (size_t)j * 3 * D + D + h * hd;
        const float* vrow = base + (size_t)j * 3 * D + 2 * D + h * hd;
        float lg = 0.0f;
        for (int d = 0; d < hd; ++d) lg += q[d] * krow[d];
        lg *= scale;
        float nm = fmaxf(m, lg);
        float cs = __expf(m - nm);
        float w  = __expf(lg - nm);
        ssum = ssum * cs + w;
        for (int d = 0; d < hd; ++d) o[d] = o[d] * cs + w * vrow[d];
        m = nm;
    }
    float inv = 1.0f / ssum;
    float* orow = obuf + ((size_t)b * S + qi) * D + h * hd;
    for (int d = 0; d < hd; ++d) orow[d] = o[d] * inv;
}

// fout = fin + layernorm(tmp)
__global__ void ln_resid_kernel(const float* __restrict__ tmp, const float* __restrict__ fin,
                                const float* __restrict__ g, const float* __restrict__ b,
                                float* __restrict__ fout, int M, int D)
{
    int m = blockIdx.x * blockDim.x + threadIdx.x;
    if (m >= M) return;
    const float* row = tmp + (size_t)m * D;
    float mu = 0.0f;
    for (int c = 0; c < D; ++c) mu += row[c];
    mu /= (float)D;
    float var = 0.0f;
    for (int c = 0; c < D; ++c) { float d = row[c] - mu; var += d * d; }
    var /= (float)D;
    float inv = rsqrtf(var + EPSBN);
    for (int c = 0; c < D; ++c)
        fout[(size_t)m * D + c] = fin[(size_t)m * D + c] + (row[c] - mu) * inv * g[c] + b[c];
}

__global__ void mean_kernel(const float* __restrict__ f, float* __restrict__ meanOut,
                            int S, int D, int total)
{
    int gid = blockIdx.x * blockDim.x + threadIdx.x;
    if (gid >= total) return;
    int b = gid / D, c = gid % D;
    float s = 0.0f;
    for (int j = 0; j < S; ++j) s += f[((size_t)b * S + j) * D + c];
    meanOut[gid] = s / (float)S;
}

__global__ void cat_kernel(const float* __restrict__ f, const float* __restrict__ mean,
                           float* __restrict__ cat, int S, int D, size_t total)
{
    size_t gid = (size_t)blockIdx.x * blockDim.x + threadIdx.x;
    if (gid >= total) return;
    int K2 = 2 * D;
    size_t m = gid / K2;
    int k = (int)(gid % K2);
    int b = (int)(m / S);
    int kf = (k < D) ? k : D - 1;                 // clamped
    int km = (k >= D) ? (k - D) : 0;              // clamped
    float a = f[m * D + kf];
    float c = mean[(size_t)b * D + km];
    cat[gid] = (k < D) ? a : c;
}

// =====================================================================
// Host: full forward pipeline
// =====================================================================
extern "C" void kernel_launch(void* const* d_in, const int* in_sizes, int n_in,
                              void* d_out, int out_size, void* d_ws, size_t ws_size,
                              hipStream_t stream)
{
    (void)in_sizes; (void)n_in; (void)out_size; (void)ws_size;
    const int B = 64, N0 = 4096;
    const float* xyz   = (const float*)d_in[0];
    const float* feats = (const float*)d_in[1];
    const float* fc0_w = (const float*)d_in[2];
    const float* fc0_b = (const float*)d_in[3];
    const float* bn0_g = (const float*)d_in[4];
    const float* bn0_b = (const float*)d_in[5];
    const float* bn0_m = (const float*)d_in[6];
    const float* bn0_v = (const float*)d_in[7];
    const float *sa_g[4], *sa_b[4], *sa_m[4], *sa_v[4], *sa_w[4];
    for (int i = 0; i < 4; ++i) {
        int base = 8 + 5 * i;
        sa_g[i] = (const float*)d_in[base + 0];
        sa_b[i] = (const float*)d_in[base + 1];
        sa_m[i] = (const float*)d_in[base + 2];
        sa_v[i] = (const float*)d_in[base + 3];
        sa_w[i] = (const float*)d_in[base + 4];
    }
    const float *att_in_w[3], *att_in_b[3], *att_out_w[3], *att_out_b[3], *att_ln_g[3], *att_ln_b[3];
    for (int i = 0; i < 3; ++i) {
        int base = 28 + 6 * i;
        att_in_w[i]  = (const float*)d_in[base + 0];
        att_in_b[i]  = (const float*)d_in[base + 1];
        att_out_w[i] = (const float*)d_in[base + 2];
        att_out_b[i] = (const float*)d_in[base + 3];
        att_ln_g[i]  = (const float*)d_in[base + 4];
        att_ln_b[i]  = (const float*)d_in[base + 5];
    }
    const float *glob_w[3], *glob_b[3];
    for (int i = 0; i < 3; ++i) {
        glob_w[i] = (const float*)d_in[46 + 2 * i];
        glob_b[i] = (const float*)d_in[47 + 2 * i];
    }

    // ---- workspace carving (floats) ----
    float* wsp = (float*)d_ws;
    size_t off = 0;
    auto F = [&](size_t n) { float* r = wsp + off; off += n; return r; };
    float* f0    = F((size_t)B * 4096 * 16);
    float* fsa1  = F((size_t)B * 512 * 64);
    float* fsa2  = F((size_t)B * 128 * 128);
    float* fatt2 = F((size_t)B * 128 * 128);
    float* fout2 = F((size_t)B * 128 * 128);
    float* fsa3  = F((size_t)B * 32 * 256);
    float* fatt3 = F((size_t)B * 32 * 256);
    float* fout3 = F((size_t)B * 32 * 256);
    float* fsa4  = F((size_t)B * 16 * 256);
    float* fatt4 = F((size_t)B * 16 * 256);
    float* fout4 = F((size_t)B * 16 * 256);
    float* pc1   = F((size_t)B * 512 * 3);
    float* pc2   = F((size_t)B * 128 * 3);
    float* pc3   = F((size_t)B * 32 * 3);
    float* pc4   = F((size_t)B * 16 * 3);
    float* dist  = F((size_t)B * 4096);
    int*   fpsix = (int*)F((size_t)B * 512);
    int*   ballix= (int*)F((size_t)B * 512 * 32);
    float* qkvb  = F((size_t)B * 128 * 384);
    float* obuf  = F((size_t)B * 128 * 128);
    float* tmpb  = F((size_t)B * 128 * 128);
    float* catb  = F((size_t)B * 128 * 256);
    float* meanb = F((size_t)B * 256);

    auto sa_stage = [&](const float* pc, int Nprev, const float* fin, int ci, int isa,
                        float* pcOut, int npoint, int nsample, float r2, int co, float* fsaOut) {
        fps_kernel<<<B, 256, 0, stream>>>(pc, Nprev, npoint, dist, fpsix);
        int tot = B * npoint;
        gather_xyz_kernel<<<(tot + 255) / 256, 256, 0, stream>>>(pc, fpsix, pcOut, Nprev, npoint, tot);
        ball_query_kernel<<<(tot + 7) / 8, 256, 0, stream>>>(pc, pcOut, Nprev, npoint, nsample, r2, ballix, tot);
        int nw = (nsample / 16) * (co / 16);
        sa_conv_kernel<<<tot, 32 * nw, 0, stream>>>(pc, pcOut, fin, ballix,
            sa_w[isa], sa_g[isa], sa_b[isa], sa_m[isa], sa_v[isa],
            fsaOut, Nprev, npoint, nsample, ci, co);
    };

    auto att_glob = [&](int ai, float* fsa, float* fatt, float* foutp, int S, int D) {
        int M = B * S, hd = D / 8;
        // qkv = x @ in_w.T + in_b
        gemm_wmma_kernel<1, false, 0><<<dim3(3 * D / 16, M / 64), 128, 0, stream>>>(
            fsa, att_in_w[ai], att_in_b[ai], nullptr, qkvb, M, D, 3 * D);
        int tot = B * 8 * S;
        attn_kernel<<<(tot + 255) / 256, 256, 0, stream>>>(qkvb, obuf, S, D, hd, tot);
        // tmp = x + o @ out_w.T + out_b
        gemm_wmma_kernel<1, true, 0><<<dim3(D / 16, M / 64), 128, 0, stream>>>(
            obuf, att_out_w[ai], att_out_b[ai], fsa, tmpb, M, D, D);
        ln_resid_kernel<<<(M + 255) / 256, 256, 0, stream>>>(tmpb, fsa, att_ln_g[ai], att_ln_b[ai], fatt, M, D);
        int mt = B * D;
        mean_kernel<<<(mt + 255) / 256, 256, 0, stream>>>(fatt, meanb, S, D, mt);
        size_t ct = (size_t)M * 2 * D;
        cat_kernel<<<(unsigned)((ct + 255) / 256), 256, 0, stream>>>(fatt, meanb, catb, S, D, ct);
        // f = leaky_relu(cat @ glob_w + glob_b)
        gemm_wmma_kernel<0, false, 1><<<dim3(D / 16, M / 64), 128, 0, stream>>>(
            catb, glob_w[ai], glob_b[ai], nullptr, foutp, M, 2 * D, D);
    };

    // fc0
    fc0_kernel<<<(B * N0 + 255) / 256, 256, 0, stream>>>(feats, fc0_w, fc0_b,
        bn0_g, bn0_b, bn0_m, bn0_v, f0, B * N0);

    // Stage 1: 4096 -> 512, r=0.2, ns=32, ci=16, co=64 (no attention)
    sa_stage(xyz, 4096, f0, 16, 0, pc1, 512, 32, 0.04f, 64, fsa1);
    // Stage 2: 512 -> 128, r=0.4, ns=32, ci=64, co=128
    sa_stage(pc1, 512, fsa1, 64, 1, pc2, 128, 32, 0.16f, 128, fsa2);
    att_glob(0, fsa2, fatt2, fout2, 128, 128);
    // Stage 3: 128 -> 32, r=0.8, ns=16, ci=128, co=256
    sa_stage(pc2, 128, fout2, 128, 2, pc3, 32, 16, 0.64f, 256, fsa3);
    att_glob(1, fsa3, fatt3, fout3, 32, 256);
    // Stage 4: 32 -> 16, r=1.2, ns=16, ci=256, co=256
    sa_stage(pc3, 32, fout3, 256, 3, pc4, 16, 16, 1.44f, 256, fsa4);
    att_glob(2, fsa4, fatt4, fout4, 16, 256);

    // ---- assemble outputs: (f, xyz, pc1, pc2, pc3, pc4) flat ----
    float* out = (float*)d_out;
    size_t o = 0;
    hipMemcpyAsync(out + o, fout4, sizeof(float) * (size_t)B * 16 * 256, hipMemcpyDeviceToDevice, stream); o += (size_t)B * 16 * 256;
    hipMemcpyAsync(out + o, xyz,  sizeof(float) * (size_t)B * 4096 * 3, hipMemcpyDeviceToDevice, stream); o += (size_t)B * 4096 * 3;
    hipMemcpyAsync(out + o, pc1,  sizeof(float) * (size_t)B * 512 * 3,  hipMemcpyDeviceToDevice, stream); o += (size_t)B * 512 * 3;
    hipMemcpyAsync(out + o, pc2,  sizeof(float) * (size_t)B * 128 * 3,  hipMemcpyDeviceToDevice, stream); o += (size_t)B * 128 * 3;
    hipMemcpyAsync(out + o, pc3,  sizeof(float) * (size_t)B * 32 * 3,   hipMemcpyDeviceToDevice, stream); o += (size_t)B * 32 * 3;
    hipMemcpyAsync(out + o, pc4,  sizeof(float) * (size_t)B * 16 * 3,   hipMemcpyDeviceToDevice, stream);
}
